// MergeZoom_75874892251838
// MI455X (gfx1250) — compile-verified, matrix operand order checked by reading, not run on hardware
//
#include <hip/hip_runtime.h>
#include <hip/hip_bf16.h>

#define MZ_H 1024
#define MZ_W 1024
#define MZ_NS 16

typedef __attribute__((ext_vector_type(16))) _Float16 v16h;
typedef __attribute__((ext_vector_type(8)))  float    v8f;

// d_ws int layout:
// [0..15]   rowMin   [16..31] rowMax
// [32..47]  colMin   [48..63] colMax
// [64..79]  hstart   [80..95] wstart   [96..111] csize

__global__ void mz_init(int* __restrict__ ws) {
  int t = threadIdx.x;
  if (t < MZ_NS) {
    ws[t]      = 0x7FFFFFFF;  // rowMin
    ws[16 + t] = -1;          // rowMax
    ws[32 + t] = 0x7FFFFFFF;  // colMin
    ws[48 + t] = -1;          // colMax
  }
}

__device__ __forceinline__ _Float16 mz_th(float v) {
  return (v >= 0.5f) ? (_Float16)1.0f : (_Float16)0.0f;
}

// Row-sum reduction on the matrix pipe: rowsums = thresh(mask_tile) x ones.
// One wave handles 16 rows x 1024 cols; 8 waves/block -> block covers 128 rows.
__global__ void mz_rows(const float* __restrict__ masks, int* __restrict__ ws) {
  const int s    = blockIdx.y;
  const int tid  = threadIdx.x;
  const int wave = tid >> 5;
  const int lane = tid & 31;
  const int M    = lane & 15;
  const int hi   = lane >> 4;
  const int rowBase = blockIdx.x * 128 + wave * 16;
  const int r = rowBase + M;
  const float* mrow = masks + ((size_t)s * MZ_H + r) * MZ_W;

  v8f acc = {};
  v16h bones;
  #pragma unroll
  for (int i = 0; i < 16; ++i) bones[i] = (_Float16)1.0f;

  // ISA 16-bit A 16x32 layout: lanes 0-15 hold K{0..7,16..23}, lanes 16-31 hold K{8..15,24..31}
  for (int kb = 0; kb < MZ_W; kb += 32) {
    const int c0 = kb + hi * 8;
    const int c1 = c0 + 16;
    float4 p0 = *(const float4*)(mrow + c0);
    float4 p1 = *(const float4*)(mrow + c0 + 4);
    float4 q0 = *(const float4*)(mrow + c1);
    float4 q1 = *(const float4*)(mrow + c1 + 4);
    v16h a;
    a[0]  = mz_th(p0.x); a[1]  = mz_th(p0.y); a[2]  = mz_th(p0.z); a[3]  = mz_th(p0.w);
    a[4]  = mz_th(p1.x); a[5]  = mz_th(p1.y); a[6]  = mz_th(p1.z); a[7]  = mz_th(p1.w);
    a[8]  = mz_th(q0.x); a[9]  = mz_th(q0.y); a[10] = mz_th(q0.z); a[11] = mz_th(q0.w);
    a[12] = mz_th(q1.x); a[13] = mz_th(q1.y); a[14] = mz_th(q1.z); a[15] = mz_th(q1.w);
    acc = __builtin_amdgcn_wmma_f32_16x16x32_f16(
        false, a, false, bones, (short)0, acc, false, false);
  }

  // D layout: VGPR j = row (base + j) on lanes 0-15, row (base + 8 + j) on lanes 16-31,
  // replicated across N. Lanes 0 and 16 harvest their 8 row sums.
  if (M == 0) {
    const int base = rowBase + hi * 8;
    int mn = 0x7FFFFFFF, mx = -1;
    #pragma unroll
    for (int j = 0; j < 8; ++j) {
      if (acc[j] > 0.5f) { const int rr = base + j; mn = min(mn, rr); mx = max(mx, rr); }
    }
    if (mx >= 0) { atomicMin(&ws[s], mn); atomicMax(&ws[16 + s], mx); }
  }
}

// Column presence: fully coalesced column-parallel stream with LDS reduction.
__global__ void mz_cols(const float* __restrict__ masks, int* __restrict__ ws) {
  const int s   = blockIdx.y;
  const int col = blockIdx.x * blockDim.x + threadIdx.x;
  const float* base = masks + (size_t)s * MZ_H * MZ_W + col;
  bool present = false;
  for (int r = 0; r < MZ_H; ++r) {
    if ((r & 31) == 0) {
      int pr = min(r + 32, MZ_H - 1);
      __builtin_prefetch(base + (size_t)pr * MZ_W, 0, 1);  // global_prefetch_b8
    }
    present = present || (base[(size_t)r * MZ_W] >= 0.5f);
  }
  __shared__ int smin[256];
  __shared__ int smax[256];
  smin[threadIdx.x] = present ? col : 0x7FFFFFFF;
  smax[threadIdx.x] = present ? col : -1;
  __syncthreads();
  for (int off = 128; off > 0; off >>= 1) {
    if ((int)threadIdx.x < off) {
      smin[threadIdx.x] = min(smin[threadIdx.x], smin[threadIdx.x + off]);
      smax[threadIdx.x] = max(smax[threadIdx.x], smax[threadIdx.x + off]);
    }
    __syncthreads();
  }
  if (threadIdx.x == 0 && smax[0] >= 0) {
    atomicMin(&ws[32 + s], smin[0]);
    atomicMax(&ws[48 + s], smax[0]);
  }
}

__global__ void mz_finalize(int* __restrict__ ws) {
  int s = threadIdx.x;
  if (s >= MZ_NS) return;
  int rmin = ws[s], rmax = ws[16 + s], cmin = ws[32 + s], cmax = ws[48 + s];
  int xl, xr, yl, yr;
  if (rmax < 0) { xl = 0; xr = MZ_H; } else { xl = max(rmin - 1, 0); xr = rmax + 1; }
  if (cmax < 0) { yl = 0; yr = MZ_W; } else { yl = max(cmin - 1, 0); yr = cmax + 1; }
  int h = xr - xl, w = yr - yl;
  int c = min(h, w);
  ws[64 + s] = xl + (h - c) / 2;  // hstart
  ws[80 + s] = yl + (w - c) / 2;  // wstart
  ws[96 + s] = c;                 // square crop side
}

__global__ void mz_resize(const float* __restrict__ masks, const float* __restrict__ images,
                          const int* __restrict__ ws, float* __restrict__ out) {
  const int s   = blockIdx.y;
  const int idx = blockIdx.x * blockDim.x + threadIdx.x;
  const int oy  = idx >> 10;
  const int ox  = idx & (MZ_W - 1);
  const int hstart = ws[64 + s];
  const int wstart = ws[80 + s];
  const int cs     = ws[96 + s];

  const float cf = (float)cs;
  const float sc = cf * (1.0f / (float)MZ_H);
  float ys = fminf(fmaxf(((float)oy + 0.5f) * sc - 0.5f, 0.0f), cf - 1.0f) + (float)hstart;
  float xs = fminf(fmaxf(((float)ox + 0.5f) * sc - 0.5f, 0.0f), cf - 1.0f) + (float)wstart;
  int y0 = (int)floorf(ys);
  int x0 = (int)floorf(xs);
  int y1 = min(y0 + 1, hstart + cs - 1);
  int x1 = min(x0 + 1, wstart + cs - 1);
  float wy = ys - (float)y0;
  float wx = xs - (float)x0;

  const float* mbase = masks  + (size_t)s * MZ_H * MZ_W;
  const float* ibase = images + (size_t)s * MZ_H * MZ_W * 3;

  auto tap = [&](int y, int x, float* rgb) {
    float mk = (mbase[(size_t)y * MZ_W + x] >= 0.5f) ? 1.0f : 0.0f;
    const float* p = ibase + ((size_t)y * MZ_W + x) * 3;
    rgb[0] = p[0] * mk; rgb[1] = p[1] * mk; rgb[2] = p[2] * mk;
  };
  float t00[3], t01[3], t10[3], t11[3];
  tap(y0, x0, t00); tap(y0, x1, t01); tap(y1, x0, t10); tap(y1, x1, t11);

  float* o = out + (((size_t)s * MZ_H + oy) * MZ_W + ox) * 3;
  #pragma unroll
  for (int ch = 0; ch < 3; ++ch) {
    float top = (1.0f - wx) * t00[ch] + wx * t01[ch];
    float bot = (1.0f - wx) * t10[ch] + wx * t11[ch];
    o[ch] = (1.0f - wy) * top + wy * bot;
  }
}

extern "C" void kernel_launch(void* const* d_in, const int* in_sizes, int n_in,
                              void* d_out, int out_size, void* d_ws, size_t ws_size,
                              hipStream_t stream) {
  (void)in_sizes; (void)n_in; (void)out_size; (void)ws_size;
  const float* masks  = (const float*)d_in[0];
  const float* images = (const float*)d_in[1];
  float* out = (float*)d_out;
  int*   ws  = (int*)d_ws;

  mz_init    <<<1, 64, 0, stream>>>(ws);
  mz_rows    <<<dim3(8, MZ_NS), 256, 0, stream>>>(masks, ws);
  mz_cols    <<<dim3(MZ_W / 256, MZ_NS), 256, 0, stream>>>(masks, ws);
  mz_finalize<<<1, 32, 0, stream>>>(ws);
  mz_resize  <<<dim3((MZ_H * MZ_W) / 256, MZ_NS), 256, 0, stream>>>(masks, images, ws, out);
}
